// GraphAttention_33371895889948
// MI455X (gfx1250) — matrix operand order repaired and checked by reading, not compiled, not run on hardware
//
#include <hip/hip_runtime.h>

typedef __attribute__((ext_vector_type(16))) __bf16 v16bf;
typedef __attribute__((ext_vector_type(8)))  float  v8f;

#define NN   4096
#define FF   1024
#define UU   128
#define HH   8

__device__ __forceinline__ unsigned short f2bf(float f) {
    unsigned int u = __float_as_uint(f);
    unsigned int r = (u + 0x7FFFu + ((u >> 16) & 1u)) >> 16;   // RNE
    return (unsigned short)r;
}
__device__ __forceinline__ float bf2f(unsigned short s) {
    return __uint_as_float(((unsigned int)s) << 16);
}

union Frag16 {
    uint4 q[2];
    v16bf v;
    unsigned short u[16];
};

// ---------------------------------------------------------------------------
// K1: cast X [N,F] f32 -> bf16 row-major
__global__ void cast_x_kernel(const float* __restrict__ X, unsigned short* __restrict__ Xb) {
    int i = blockIdx.x * blockDim.x + threadIdx.x;
    if (i < NN * FF) Xb[i] = f2bf(X[i]);
}

// K1b: pack W [H,F,U] f32 -> bf16 in WMMA B-fragment order:
//      Wp[h][kb(32)][ub(8)][lane(32)][e(16)],  K = kb*32 + e + (lane>=16 ? 16 : 0),
//      u = ub*16 + (lane&15)
__global__ void pack_w_kernel(const float* __restrict__ W, unsigned short* __restrict__ Wp) {
    int i = blockIdx.x * blockDim.x + threadIdx.x;
    if (i >= HH * FF * UU) return;
    int e    = i & 15;
    int lane = (i >> 4) & 31;
    int ub   = (i >> 9) & 7;
    int kb   = (i >> 12) & 31;
    int h    = i >> 17;
    int k = kb * 32 + e + ((lane >= 16) ? 16 : 0);
    int u = ub * 16 + (lane & 15);
    Wp[i] = f2bf(W[(h * FF + k) * UU + u]);
}

// ---------------------------------------------------------------------------
// K2: projection GEMM  feat[h] = X (bf16) @ W[h] (bf16), f32 accum via WMMA.
// One wave computes one 16x16 tile; epilogue scatters into the B-fragment
// packed feat buffer:
//   featB[h][jb(128)][ub(8)][lane(32)][e(16)], K(=j within 32-block) = e + 16*(lane>=16)
__global__ __launch_bounds__(128)
void proj_gemm_kernel(const unsigned short* __restrict__ Xb,
                      const unsigned short* __restrict__ Wp,
                      unsigned short* __restrict__ featB) {
    int mtile = blockIdx.x;                 // 0..255 (16-row tile of N)
    int wid   = threadIdx.x >> 5;
    int lane  = threadIdx.x & 31;
    int ct    = blockIdx.y * 4 + wid;       // 0..63 = h*8 + ub
    int h     = ct >> 3;
    int ub    = ct & 7;
    int half  = lane >> 4;
    int rowlo = lane & 15;

    const unsigned short* xrow = Xb + (size_t)(mtile * 16 + rowlo) * FF;
    const unsigned short* wp   = Wp + ((size_t)(h * 32) * 8 + ub) * 512 + lane * 16;

    v8f acc = {};
    for (int kb = 0; kb < 32; ++kb) {
        Frag16 a, b;
        int kbase = kb * 32 + half * 16;    // A: lanes0-15 K 0..15; lanes16-31 K 16..31
        a.q[0] = *(const uint4*)(xrow + kbase);
        a.q[1] = *(const uint4*)(xrow + kbase + 8);
        const unsigned short* wkb = wp + (size_t)kb * 4096;   // 8*512 shorts per kb
        b.q[0] = ((const uint4*)wkb)[0];
        b.q[1] = ((const uint4*)wkb)[1];
        acc = __builtin_amdgcn_wmma_f32_16x16x32_bf16(false, a.v, false, b.v,
                                                      (short)0, acc, false, false);
    }
    // D layout: VGPR r -> n = mtile*16 + r + half*8, u = ub*16 + (lane&15)
#pragma unroll
    for (int r = 0; r < 8; ++r) {
        int n  = mtile * 16 + r + half * 8;
        int jb = n >> 5;
        int K  = n & 31;
        int lb = (lane & 15) | ((K >> 4) << 4);
        int e  = K & 15;
        featB[((((size_t)h * 128 + jb) * 8 + ub) * 32 + lb) * 16 + e] = f2bf(acc[r]);
    }
}

// ---------------------------------------------------------------------------
// K3: s[h,n] = feat[h,n,:].a_self ; t[h,n] = feat[h,n,:].a_neigh
__global__ void st_kernel(const unsigned short* __restrict__ featB,
                          const float* __restrict__ a_self,
                          const float* __restrict__ a_neigh,
                          float* __restrict__ s, float* __restrict__ t) {
    int i = blockIdx.x * blockDim.x + threadIdx.x;     // h*N + n
    if (i >= HH * NN) return;
    int h = i >> 12;
    int n = i & (NN - 1);
    int jb = n >> 5, K = n & 31;
    int hb = (K >> 4) << 4, e = K & 15;
    float ss = 0.f, tt = 0.f;
    for (int u = 0; u < UU; ++u) {
        int ub = u >> 4;
        int lb = (u & 15) | hb;
        float f = bf2f(featB[((((size_t)h * 128 + jb) * 8 + ub) * 32 + lb) * 16 + e]);
        ss += f * a_self[h * UU + u];
        tt += f * a_neigh[h * UU + u];
    }
    s[i] = ss;
    t[i] = tt;
}

// ---------------------------------------------------------------------------
// K4: fused masked softmax + aggregation, split-j flash style.
// Workgroup = (16-row tile, head), 4 waves. Wave w processes j-tiles
// [w*32, w*32+32); each wave owns ALL 128 U columns (8 WMMA acc tiles), so the
// softmax VALU work is never duplicated. Final LDS merge combines the four
// partial (m, l, acc) triples.
__global__ __launch_bounds__(128)
void attn_kernel(const float* __restrict__ adj,
                 const unsigned short* __restrict__ featB,
                 const float* __restrict__ s, const float* __restrict__ t,
                 const float* __restrict__ bias, float* __restrict__ out) {
    __shared__ float sh_m[4][16];
    __shared__ float sh_l[4][16];
    __shared__ float sh_acc[4][16][UU];     // 32 KB

    const int mtile = blockIdx.x;           // 0..255
    const int h     = blockIdx.y;           // 0..7
    const int w     = threadIdx.x >> 5;     // wave id = j partition
    const int lane  = threadIdx.x & 31;
    const int half  = lane >> 4;
    const int rowlo = lane & 15;
    const int i     = mtile * 16 + rowlo;

    const float  si    = s[h * NN + i];
    const float* arow  = adj + (size_t)i * NN;
    const float* trow  = t + (size_t)h * NN;
    const float  LOG2E = 1.44269504088896f;

    v8f acc[8] = {};
    float m_run = -INFINITY, l_run = 0.f;

    for (int jb = w * 32; jb < w * 32 + 32; ++jb) {
        const int jbase = jb * 32 + half * 16;
        // adjacency + t for this lane's 16 A-fragment K slots
        float4 a0 = *(const float4*)(arow + jbase);
        float4 a1 = *(const float4*)(arow + jbase + 4);
        float4 a2 = *(const float4*)(arow + jbase + 8);
        float4 a3 = *(const float4*)(arow + jbase + 12);
        float4 t0 = *(const float4*)(trow + jbase);
        float4 t1 = *(const float4*)(trow + jbase + 4);
        float4 t2 = *(const float4*)(trow + jbase + 8);
        float4 t3 = *(const float4*)(trow + jbase + 12);
        float av[16] = {a0.x, a0.y, a0.z, a0.w, a1.x, a1.y, a1.z, a1.w,
                        a2.x, a2.y, a2.z, a2.w, a3.x, a3.y, a3.z, a3.w};
        float tv[16] = {t0.x, t0.y, t0.z, t0.w, t1.x, t1.y, t1.z, t1.w,
                        t2.x, t2.y, t2.z, t2.w, t3.x, t3.y, t3.z, t3.w};

        float p[16];
        float mloc = -INFINITY;
#pragma unroll
        for (int e = 0; e < 16; ++e) {
            float x = si + tv[e];
            x = fmaxf(x, 0.2f * x);                       // leaky_relu
            x = fmaf(1e10f, av[e], x - 1e10f);            // += -1e10*(1-a)
            p[e] = x;
            mloc = fmaxf(mloc, x);
        }
        mloc = fmaxf(mloc, __shfl_xor(mloc, 16, 32));     // partner lane = same row
        const float m_new = fmaxf(m_run, mloc);
        const float mb    = m_new * LOG2E;

        Frag16 pa;
        float lloc = 0.f;
#pragma unroll
        for (int e = 0; e < 16; ++e) {
            float pe = exp2f(fmaf(p[e], LOG2E, -mb));
            lloc += pe;
            pa.v[e] = (__bf16)pe;                         // hw cvt
        }
        lloc += __shfl_xor(lloc, 16, 32);

        if (__any(mloc > m_run)) {                        // some row max changed
            const float scale = exp2f((m_run - m_new) * LOG2E);
            l_run = l_run * scale + lloc;
#pragma unroll
            for (int r = 0; r < 8; ++r) {
                float sc = __shfl(scale, r + half * 8, 32);   // scale of row r+half*8
#pragma unroll
                for (int ub = 0; ub < 8; ++ub) acc[ub][r] *= sc;
            }
        } else {
            l_run += lloc;                                // all scales == 1
        }
        m_run = m_new;

        const unsigned short* fb = featB + (((size_t)h * 128 + jb) * 8) * 512 + lane * 16;
#pragma unroll
        for (int ub = 0; ub < 8; ++ub) {
            Frag16 b;
            const unsigned short* f0 = fb + (size_t)ub * 512;
            b.q[0] = ((const uint4*)f0)[0];
            b.q[1] = ((const uint4*)f0)[1];
            acc[ub] = __builtin_amdgcn_wmma_f32_16x16x32_bf16(false, pa.v, false, b.v,
                                                              (short)0, acc[ub], false, false);
        }
        if (jb < NN / 32 - 1) __builtin_prefetch(arow + (jb + 1) * 32, 0, 1);
    }

    // ---- merge the 4 j-partitions through LDS ----
    if (lane < 16) { sh_m[w][lane] = m_run; sh_l[w][lane] = l_run; }
#pragma unroll
    for (int ub = 0; ub < 8; ++ub)
#pragma unroll
        for (int r = 0; r < 8; ++r)
            sh_acc[w][r + half * 8][ub * 16 + rowlo] = acc[ub][r];
    __syncthreads();

    // wave w finalizes U columns [w*32, w*32+32)
    const int ubA = w * 2, ubB = w * 2 + 1;
#pragma unroll
    for (int r = 0; r < 8; ++r) {
        const int row = r + half * 8;
        float m0 = sh_m[0][row], m1 = sh_m[1][row], m2 = sh_m[2][row], m3 = sh_m[3][row];
        float M  = fmaxf(fmaxf(m0, m1), fmaxf(m2, m3));
        float e0 = exp2f((m0 - M) * LOG2E), e1 = exp2f((m1 - M) * LOG2E);
        float e2 = exp2f((m2 - M) * LOG2E), e3 = exp2f((m3 - M) * LOG2E);
        float L = sh_l[0][row] * e0 + sh_l[1][row] * e1 +
                  sh_l[2][row] * e2 + sh_l[3][row] * e3;
        float invL = 1.f / L;
        const int n = mtile * 16 + row;
#pragma unroll
        for (int q = 0; q < 2; ++q) {
            int ub  = (q == 0) ? ubA : ubB;
            int col = ub * 16 + rowlo;
            float v = sh_acc[0][row][col] * e0 + sh_acc[1][row][col] * e1 +
                      sh_acc[2][row][col] * e2 + sh_acc[3][row][col] * e3;
            float o = fmaxf(fmaf(v, invL, bias[h * UU + col]), 0.f);
            out[(size_t)n * (HH * UU) + h * UU + col] = o;
        }
    }
}

// ---------------------------------------------------------------------------
extern "C" void kernel_launch(void* const* d_in, const int* in_sizes, int n_in,
                              void* d_out, int out_size, void* d_ws, size_t ws_size,
                              hipStream_t stream) {
    const float* X       = (const float*)d_in[0];   // [1,N,F]
    const float* A       = (const float*)d_in[1];   // [1,N,N]
    const float* W       = (const float*)d_in[2];   // [H,F,U]
    const float* a_self  = (const float*)d_in[3];   // [H,U]
    const float* a_neigh = (const float*)d_in[4];   // [H,U]
    const float* bias    = (const float*)d_in[5];   // [H,U]
    float* out = (float*)d_out;

    char* ws = (char*)d_ws;
    const size_t XB_BYTES = (size_t)NN * FF * 2;          // 8 MB
    const size_t WP_BYTES = (size_t)HH * FF * UU * 2;     // 2 MB
    const size_t FB_BYTES = (size_t)HH * NN * UU * 2;     // 8 MB
    const size_t S_BYTES  = (size_t)HH * NN * 4;          // 128 KB
    unsigned short* Xb    = (unsigned short*)ws;
    unsigned short* Wp    = (unsigned short*)(ws + XB_BYTES);
    unsigned short* featB = (unsigned short*)(ws + XB_BYTES + WP_BYTES);
    float* s = (float*)(ws + XB_BYTES + WP_BYTES + FB_BYTES);
    float* t = (float*)(ws + XB_BYTES + WP_BYTES + FB_BYTES + S_BYTES);

    cast_x_kernel<<<(NN * FF + 255) / 256, 256, 0, stream>>>(X, Xb);
    pack_w_kernel<<<(HH * FF * UU + 255) / 256, 256, 0, stream>>>(W, Wp);
    proj_gemm_kernel<<<dim3(NN / 16, 16), 128, 0, stream>>>(Xb, Wp, featB);
    st_kernel<<<(HH * NN + 255) / 256, 256, 0, stream>>>(featB, a_self, a_neigh, s, t);
    attn_kernel<<<dim3(NN / 16, HH), 128, 0, stream>>>(A, featB, s, t, bias, out);
}
